// GraphDiffusionBlock_1975684956300
// MI455X (gfx1250) — matrix operand order repaired and checked by reading, not compiled, not run on hardware
//
#include <hip/hip_runtime.h>
#include <hip/hip_bf16.h>

#define HEADS 4
#define LN_EPS 1e-5f
#define NEG_SLOPE 0.2f

typedef _Float16 f16;
typedef _Float16 v16h __attribute__((ext_vector_type(16)));
typedef _Float16 v8h  __attribute__((ext_vector_type(8)));
typedef _Float16 v4h  __attribute__((ext_vector_type(4)));
typedef float    v8f  __attribute__((ext_vector_type(8)));

// Monotonic float<->uint encoding so we can use hardware u32 atomic max.
__device__ __forceinline__ unsigned f2ord(float f) {
    unsigned u = __float_as_uint(f);
    return (u & 0x80000000u) ? ~u : (u | 0x80000000u);
}
__device__ __forceinline__ float ord2f(unsigned u) {
    return __uint_as_float((u & 0x80000000u) ? (u ^ 0x80000000u) : ~u);
}
__device__ __forceinline__ float lrelu(float v) { return v > 0.f ? v : NEG_SLOPE * v; }

// ---------------- fp32 -> fp16 conversion (float4 -> 4x f16 per thread) ----------------
__global__ void k_cvt_f16(const float* __restrict__ in, f16* __restrict__ out, int n4) {
    int i = blockIdx.x * blockDim.x + threadIdx.x;
    if (i >= n4) return;
    float4 v = ((const float4*)in)[i];
    v4h o;
    o[0] = (f16)v.x; o[1] = (f16)v.y; o[2] = (f16)v.z; o[3] = (f16)v.w;
    ((v4h*)out)[i] = o;
}

// ---------------- h = x @ W^T via WMMA -------------------
// xh: [N,128] f16 row-major; wh: [256,128] f16 row-major (W as-is).
// hfeat: [N,256] f32. Each wave computes one 16x16 tile; K=128 -> 4 WMMA steps.
__global__ void k_gemm_wmma(const f16* __restrict__ xh, const f16* __restrict__ wh,
                            float* __restrict__ hfeat, int N) {
    const int lane = threadIdx.x & 31;
    const int wave = threadIdx.x >> 5;
    const int hi   = lane >> 4;   // lane half
    const int lo   = lane & 15;
    const int rowbase = blockIdx.x * 16;
    const int colbase = (blockIdx.y * 8 + wave) * 16;   // 0..240

    int arow = rowbase + lo;
    if (arow >= N) arow = N - 1;                        // clamp for ragged tail
    const int ncol = colbase + lo;                      // 0..255, always valid

    v8f acc = {};
#pragma unroll
    for (int kk = 0; kk < 128; kk += 32) {
        // A 16x32 f16: halfs 0..7 -> K = kk+hi*8 .. +7 ; halfs 8..15 -> K = kk+16+hi*8 ..
        const f16* ap = xh + (size_t)arow * 128 + kk + hi * 8;
        v8h a0 = *(const v8h*)(ap);
        v8h a1 = *(const v8h*)(ap + 16);
        // B 32x16 f16: lane col = lo, halfs 0..7 -> K = kk+hi*8 .. +7 ; halfs 8..15 -> K = kk+16+hi*8 ..
        const f16* bp = wh + (size_t)ncol * 128 + kk + hi * 8;
        v8h b0 = *(const v8h*)(bp);
        v8h b1 = *(const v8h*)(bp + 16);
        v16h A, B;
#pragma unroll
        for (int j = 0; j < 8; ++j) {
            A[j] = a0[j]; A[j + 8] = a1[j];
            B[j] = b0[j]; B[j + 8] = b1[j];
        }
        acc = __builtin_amdgcn_wmma_f32_16x16x32_f16(false, A, false, B,
                                                     (short)0, acc, false, false);
    }
    // D: VGPR r -> row rowbase + r + 8*hi, col colbase + lo  (coalesced across lanes)
    if (rowbase + 16 <= N) {            // uniform fast path (N % 16 == 0 in practice)
        float* op = hfeat + (size_t)(rowbase + 8 * hi) * 256 + colbase + lo;
#pragma unroll
        for (int r = 0; r < 8; ++r) op[(size_t)r * 256] = acc[r];
    } else {
#pragma unroll
        for (int r = 0; r < 8; ++r) {
            int row = rowbase + r + 8 * hi;
            if (row < N) hfeat[(size_t)row * 256 + colbase + lo] = acc[r];
        }
    }
}

// ---------------- attention scores per (node, head) ----------------
__global__ void k_scores(const float* __restrict__ hfeat,
                         const float* __restrict__ att_src, const float* __restrict__ att_dst,
                         float* __restrict__ as_, float* __restrict__ ad_, int N) {
    int i = blockIdx.x * blockDim.x + threadIdx.x;
    if (i >= N * HEADS) return;
    int n = i >> 2, h = i & 3;
    const float4* hp = (const float4*)(hfeat + (size_t)n * 256 + h * 64);
    const float4* sp = (const float4*)(att_src + h * 64);
    const float4* dp = (const float4*)(att_dst + h * 64);
    float s = 0.f, d = 0.f;
#pragma unroll
    for (int c = 0; c < 16; ++c) {
        float4 v = hp[c], a = sp[c], b = dp[c];
        s += v.x * a.x + v.y * a.y + v.z * a.z + v.w * a.w;
        d += v.x * b.x + v.y * b.y + v.z * b.z + v.w * b.w;
    }
    as_[i] = s; ad_[i] = d;
}

// ---------------- init segment max with self-loop term ----------------
__global__ void k_max_self(const float* __restrict__ as_, const float* __restrict__ ad_,
                           unsigned* __restrict__ m_enc, int NH) {
    int i = blockIdx.x * blockDim.x + threadIdx.x;
    if (i >= NH) return;
    m_enc[i] = f2ord(lrelu(as_[i] + ad_[i]));
}

// ---------------- edge pass 1: segment max ----------------
__global__ void k_edge_max(const int* __restrict__ ei, const float* __restrict__ as_,
                           const float* __restrict__ ad_, unsigned* __restrict__ m_enc, int E) {
    int i = blockIdx.x * blockDim.x + threadIdx.x;
    if (i >= E * HEADS) return;
    int e = i >> 2, h = i & 3;
    int src = ei[e], dst = ei[E + e];
    float v = lrelu(as_[src * 4 + h] + ad_[dst * 4 + h]);
    atomicMax(&m_enc[dst * 4 + h], f2ord(v));
}

// ---------------- init denom & S with self-loop contribution (float4) ----------------
__global__ void k_self_init(const float* __restrict__ hfeat, const float* __restrict__ as_,
                            const float* __restrict__ ad_, const unsigned* __restrict__ m_enc,
                            float* __restrict__ denom, float* __restrict__ S, int N) {
    int i = blockIdx.x * blockDim.x + threadIdx.x;   // (n, 4-channel group), 64 groups/node
    if (i >= N * 64) return;
    int n = i >> 6, g = i & 63, h = g >> 4;          // g*4 = channel base, h = head
    float w = __expf(lrelu(as_[n * 4 + h] + ad_[n * 4 + h]) - ord2f(m_enc[n * 4 + h]));
    float4 v = ((const float4*)hfeat)[i];
    float4 o; o.x = w * v.x; o.y = w * v.y; o.z = w * v.z; o.w = w * v.w;
    ((float4*)S)[i] = o;
    if ((g & 15) == 0) denom[n * 4 + h] = w;
}

// ---------------- edge pass 2a: per-edge softmax weights (computed once) ----------------
__global__ void k_edge_w(const int* __restrict__ ei, const float* __restrict__ as_,
                         const float* __restrict__ ad_, const unsigned* __restrict__ m_enc,
                         float* __restrict__ wbuf, float* __restrict__ denom, int E) {
    int i = blockIdx.x * blockDim.x + threadIdx.x;
    if (i >= E * HEADS) return;
    int e = i >> 2, h = i & 3;
    int src = ei[e], dst = ei[E + e];
    float v = lrelu(as_[src * 4 + h] + ad_[dst * 4 + h]);
    float w = __expf(v - ord2f(m_enc[dst * 4 + h]));
    wbuf[i] = w;
    atomicAdd(&denom[dst * 4 + h], w);
}

// ---------------- edge pass 2b: weighted scatter-aggregate ----------------
// 64 threads/edge: thread j -> head j>>4, channel group (j&15)*4 (float4 read, 4 atomics).
__global__ void k_edge_agg(const int* __restrict__ ei, const float* __restrict__ hfeat,
                           const float* __restrict__ wbuf, float* __restrict__ S, int E) {
    int t = blockIdx.x * blockDim.x + threadIdx.x;
    int e = t >> 6;
    if (e >= E) return;
    int j = t & 63;
    int h = j >> 4, g = j & 15;
    int src = ei[e], dst = ei[E + e];
    float w = wbuf[e * 4 + h];
    float4 v = *(const float4*)(hfeat + (size_t)src * 256 + h * 64 + g * 4);
    float* sp = S + (size_t)dst * 256 + h * 64 + g * 4;
    atomicAdd(sp + 0, w * v.x);
    atomicAdd(sp + 1, w * v.y);
    atomicAdd(sp + 2, w * v.z);
    atomicAdd(sp + 3, w * v.w);
}

// ---------------- tiny time-projection GEMM: tpb[B,64] ----------------
__global__ void k_time_proj(const float* __restrict__ te, const float* __restrict__ tp_w,
                            const float* __restrict__ tp_b, float* __restrict__ tpb) {
    int b = blockIdx.x, c = threadIdx.x;
    float acc = 0.f;
#pragma unroll 8
    for (int k = 0; k < 128; ++k) {
        float s = te[b * 128 + k];
        float si = s / (1.f + __expf(-s));    // SiLU
        acc += si * tp_w[c * 128 + k];
    }
    tpb[b * 64 + c] = acc + tp_b[c];
}

// ---------------- finalize: normalize, head-mean, +tp, LayerNorm, SiLU ----------------
// One wave per node; lane handles channels {lane, lane+32}.
__global__ void k_finalize(const float* __restrict__ S, const float* __restrict__ denom,
                           const float* __restrict__ conv_bias, const float* __restrict__ tpb,
                           const int* __restrict__ batch,
                           const float* __restrict__ ln_g, const float* __restrict__ ln_b,
                           float* __restrict__ out, int N) {
    int n = (blockIdx.x * blockDim.x + threadIdx.x) >> 5;
    int lane = threadIdx.x & 31;
    if (n >= N) return;   // whole wave exits together
    float inv[HEADS];
#pragma unroll
    for (int h = 0; h < HEADS; ++h) inv[h] = 1.f / (denom[n * 4 + h] + 1e-16f);
    int b = batch[n];
    float v[2];
#pragma unroll
    for (int t = 0; t < 2; ++t) {
        int c = lane + 32 * t;
        float acc = 0.f;
#pragma unroll
        for (int h = 0; h < HEADS; ++h) acc += S[(size_t)n * 256 + h * 64 + c] * inv[h];
        v[t] = acc * 0.25f + conv_bias[c] + tpb[b * 64 + c];
    }
    float s = v[0] + v[1];
    float q = v[0] * v[0] + v[1] * v[1];
#pragma unroll
    for (int m = 16; m >= 1; m >>= 1) {
        s += __shfl_xor(s, m, 32);
        q += __shfl_xor(q, m, 32);
    }
    float mu  = s * (1.f / 64.f);
    float var = q * (1.f / 64.f) - mu * mu;
    float rs  = rsqrtf(var + LN_EPS);
#pragma unroll
    for (int t = 0; t < 2; ++t) {
        int c = lane + 32 * t;
        float z = (v[t] - mu) * rs * ln_g[c] + ln_b[c];
        out[(size_t)n * 64 + c] = z / (1.f + __expf(-z));   // SiLU
    }
}

extern "C" void kernel_launch(void* const* d_in, const int* in_sizes, int n_in,
                              void* d_out, int out_size, void* d_ws, size_t ws_size,
                              hipStream_t stream) {
    const float* x         = (const float*)d_in[0];
    const int*   ei        = (const int*)  d_in[1];
    const float* te        = (const float*)d_in[2];
    const int*   batch     = (const int*)  d_in[3];
    const float* W         = (const float*)d_in[4];
    const float* att_src   = (const float*)d_in[5];
    const float* att_dst   = (const float*)d_in[6];
    const float* conv_bias = (const float*)d_in[7];
    const float* tp_w      = (const float*)d_in[8];
    const float* tp_b      = (const float*)d_in[9];
    const float* ln_g      = (const float*)d_in[10];
    const float* ln_b      = (const float*)d_in[11];
    float* out = (float*)d_out;

    const int N = in_sizes[0] / 128;    // 50000
    const int E = in_sizes[1] / 2;      // 800000
    const int B = in_sizes[2] / 128;    // 16

    // Workspace partition (all buffers fully overwritten each call; ~130 MB)
    char* p = (char*)d_ws;
    auto alloc = [&](size_t bytes) -> char* {
        char* r = p;
        p += (bytes + 255) & ~(size_t)255;
        return r;
    };
    f16*      xh    = (f16*)     alloc((size_t)N * 128 * sizeof(f16));
    f16*      wh    = (f16*)     alloc((size_t)256 * 128 * sizeof(f16));
    float*    hfeat = (float*)   alloc((size_t)N * 256 * sizeof(float));
    float*    as_   = (float*)   alloc((size_t)N * HEADS * sizeof(float));
    float*    ad_   = (float*)   alloc((size_t)N * HEADS * sizeof(float));
    unsigned* m_enc = (unsigned*)alloc((size_t)N * HEADS * sizeof(unsigned));
    float*    denom = (float*)   alloc((size_t)N * HEADS * sizeof(float));
    float*    S     = (float*)   alloc((size_t)N * 256 * sizeof(float));
    float*    wbuf  = (float*)   alloc((size_t)E * HEADS * sizeof(float));
    float*    tpb   = (float*)   alloc((size_t)B * 64 * sizeof(float));

    // 1) fp32 -> fp16 for WMMA operands
    {
        int n4 = N * 128 / 4;
        k_cvt_f16<<<(n4 + 255) / 256, 256, 0, stream>>>(x, xh, n4);
        int m4 = 256 * 128 / 4;
        k_cvt_f16<<<(m4 + 255) / 256, 256, 0, stream>>>(W, wh, m4);
    }
    // 2) h = x @ W^T  (WMMA f16 -> f32)
    {
        dim3 grid((N + 15) / 16, 2);
        k_gemm_wmma<<<grid, 256, 0, stream>>>(xh, wh, hfeat, N);
    }
    // 3) attention scores
    k_scores<<<(N * HEADS + 255) / 256, 256, 0, stream>>>(hfeat, att_src, att_dst, as_, ad_, N);
    // 4) segment max (self init + edge atomics)
    k_max_self<<<(N * HEADS + 255) / 256, 256, 0, stream>>>(as_, ad_, m_enc, N * HEADS);
    k_edge_max<<<(E * HEADS + 255) / 256, 256, 0, stream>>>(ei, as_, ad_, m_enc, E);
    // 5) self contributions to denom & S
    k_self_init<<<(N * 64 + 255) / 256, 256, 0, stream>>>(hfeat, as_, ad_, m_enc, denom, S, N);
    // 6a) per-edge softmax weights + denom accumulation
    k_edge_w<<<(E * HEADS + 255) / 256, 256, 0, stream>>>(ei, as_, ad_, m_enc, wbuf, denom, E);
    // 6b) edge scatter-aggregate (64 threads/edge, float4 reads)
    {
        long long work = (long long)E * 64;
        k_edge_agg<<<(unsigned)((work + 255) / 256), 256, 0, stream>>>(ei, hfeat, wbuf, S, E);
    }
    // 7) time projection
    k_time_proj<<<B, 64, 0, stream>>>(te, tp_w, tp_b, tpb);
    // 8) finalize (one wave per node)
    k_finalize<<<(N + 7) / 8, 256, 0, stream>>>(S, denom, conv_bias, tpb, batch,
                                                ln_g, ln_b, out, N);
}
